// RuleSegmentationHead_13013750907214
// MI455X (gfx1250) — matrix operand order repaired and checked by reading, not compiled, not run on hardware
//
#include <hip/hip_runtime.h>

typedef __attribute__((ext_vector_type(16))) _Float16 v16h;
typedef __attribute__((ext_vector_type(8)))  _Float16 v8h;
typedef __attribute__((ext_vector_type(8)))  float    v8f;

#define HH   512
#define WW   512
#define HWP  (HH*WW)            // 262144
#define NIMG 4
#define CCAT 37                 // 32 + 5 concat channels
#define MIDC 16
#define CPAD 48                 // LDS channel padding for conv1 tile
#define CPB  (CPAD * 2)         // bytes per dx step in conv1 tile (96)

// workspace layout (bytes)
#define OFF_WF1   0u            // 98*32*16 f16 = 100352
#define OFF_WF2   102400u       // 25*32*16 f16 = 25600
#define OFF_RC8   131072u       // 4*512*512 u8 = 1 MiB
#define OFF_H     1310720u      // 4*16*512*512 f16 = 32 MiB
#define OFF_CORR  68419584u     // 4*4*512*512 f32 = 16 MiB

#define SHUF16(a, b) __builtin_shufflevector(a, b, 0, 1, 2, 3, 4, 5, 6, 7, \
                                             8, 9, 10, 11, 12, 13, 14, 15)

static __device__ __forceinline__ v8f wmma_f16(v16h a, v16h b, v8f c) {
    return __builtin_amdgcn_wmma_f32_16x16x32_f16(false, a, false, b,
                                                  (short)0, c, false, false);
}

// ---------------------------------------------------------------------------
// K0: pack w1/w2 into per-lane WMMA B-fragment order (f16), zero-padding K/N.
// B elem (lane,j) -> K = 8*(lane/16)+j (j<8), 16+8*(lane/16)+j-8 (j>=8), N = lane%16 within the chunk.
// ---------------------------------------------------------------------------
__global__ void k_pack(const float* __restrict__ w1, const float* __restrict__ w2,
                       _Float16* __restrict__ wf1, _Float16* __restrict__ wf2) {
    int id = blockIdx.x * 256 + threadIdx.x;
    if (id < 98 * 512) {                       // conv1: chunk = pos*2 + cc
        int chunk = id >> 9, lane = (id >> 4) & 31, j = id & 15;
        int pos = chunk >> 1, cc = chunk & 1;
        int ky = pos / 7, kx = pos % 7;
        int hl = lane >> 4, n = lane & 15;
        int c = cc * 32 + ((j < 8) ? 8 * hl + j : 16 + 8 * hl + (j - 8));         // input channel (0..63)
        float v = (c < CCAT) ? w1[((n * CCAT + c) * 7 + ky) * 7 + kx] : 0.f;
        wf1[id] = (_Float16)v;
    } else if (id < 98 * 512 + 25 * 512) {     // conv2: chunk packs 2 positions
        int id2 = id - 98 * 512;
        int chunk = id2 >> 9, lane = (id2 >> 4) & 31, j = id2 & 15;
        int hl = lane >> 4, n = lane & 15;
        int Kc = (j < 8) ? 8 * hl + j : 16 + 8 * hl + (j - 8);                  // 0..31
        int pos = chunk * 2 + (Kc >> 4);
        int c = Kc & 15;
        float v = (n < 4 && pos < 49) ? w2[((n * 16 + c) * 7 + (pos / 7)) * 7 + (pos % 7)] : 0.f;
        wf2[id2] = (_Float16)v;
    }
}

// ---------------------------------------------------------------------------
// K1: 3x3 conv 32->2 + ReLU, thresholds, rule gather.
// ---------------------------------------------------------------------------
__global__ __launch_bounds__(256) void k_prolog(
    const float* __restrict__ xi, const float* __restrict__ wi, const float* __restrict__ bi,
    const float* __restrict__ thr0, const float* __restrict__ thr1,
    const int* __restrict__ rules,
    float* __restrict__ proc_out, float* __restrict__ rcat_out,
    unsigned char* __restrict__ rc8) {
    __shared__ float sw[2 * 32 * 9];
    __shared__ float sb[2];
    for (int i = threadIdx.x; i < 576; i += 256) sw[i] = wi[i];
    if (threadIdx.x < 2) sb[threadIdx.x] = bi[threadIdx.x];
    __syncthreads();
    int p = blockIdx.x * 256 + threadIdx.x;
    if (p >= NIMG * HWP) return;
    int n = p >> 18, y = (p >> 9) & 511, x = p & 511;
    float a0 = sb[0], a1 = sb[1];
    for (int c = 0; c < 32; ++c) {
        const float* base = xi + (((size_t)(n * 32 + c)) << 18);
        #pragma unroll
        for (int ky = 0; ky < 3; ++ky) {
            int yy = y + ky - 1;
            if ((unsigned)yy >= (unsigned)HH) continue;
            #pragma unroll
            for (int kx = 0; kx < 3; ++kx) {
                int xx = x + kx - 1;
                if ((unsigned)xx >= (unsigned)WW) continue;
                float v = base[(yy << 9) + xx];
                a0 += v * sw[c * 9 + ky * 3 + kx];
                a1 += v * sw[288 + c * 9 + ky * 3 + kx];
            }
        }
    }
    a0 = fmaxf(a0, 0.f);
    a1 = fmaxf(a1, 0.f);
    int c0 = 0, c1 = 0;
    #pragma unroll
    for (int i = 0; i < 4; ++i) {
        c0 += (a0 >= thr0[i]);
        c1 += (a1 >= thr1[i]);
    }
    int rc = rules[c1 + c0 * 5];
    int yx = (y << 9) + x;
    proc_out[(size_t)(n * 2 + 0) * HWP + yx] = a0;
    proc_out[(size_t)(n * 2 + 1) * HWP + yx] = a1;
    rcat_out[p] = (float)rc;
    rc8[p] = (unsigned char)rc;
}

// ---------------------------------------------------------------------------
// K2: 7x7 conv (32 x_corr + 5 rule_act ch) -> 16 via WMMA implicit GEMM.
// 16x16 output tile / block; LDS tile [22][22][48] f16 (zeros c>=37).
// Ping-pong pipeline + incremental uniform LDS deltas (no div chains).
// ---------------------------------------------------------------------------
__global__ __launch_bounds__(256) __attribute__((amdgpu_waves_per_eu(4)))
void k_conv1(const float* __restrict__ xc, const float* __restrict__ ae,
             const unsigned char* __restrict__ rc8, const _Float16* __restrict__ wf1,
             const float* __restrict__ b1, _Float16* __restrict__ hbuf) {
    __shared__ __align__(16) _Float16 tile[22 * 22 * CPAD];
    int bx = blockIdx.x;
    int tx0 = (bx & 31) << 4;
    int ty0 = ((bx >> 5) & 31) << 4;
    int n = bx >> 10;
    // ---- stage halo tile (f32->f16 + rule_act materialize + zero pad) ----
    for (int flat = threadIdx.x; flat < CPAD * 484; flat += 256) {
        int c = flat / 484;
        int r = flat - c * 484;
        int dy = r / 22, dx = r - dy * 22;
        int gy = ty0 + dy - 3, gx = tx0 + dx - 3;
        float v = 0.f;
        if ((unsigned)gy < (unsigned)HH && (unsigned)gx < (unsigned)WW) {
            if (c < 32)
                v = xc[(((size_t)(n * 32 + c)) << 18) + (gy << 9) + gx];
            else if (c < CCAT) {
                int rc = rc8[(((size_t)n) << 18) + (gy << 9) + gx];
                v = ae[rc * 5 + (c - 32)];
            }
        }
        tile[(dy * 22 + dx) * CPAD + c] = (_Float16)v;
    }
    __syncthreads();
    int wave = threadIdx.x >> 5;
    int lane = threadIdx.x & 31;
    int hl = lane >> 4, m = lane & 15;     // A: pixel M = m ; B/D: channel N = m
    float bias = b1[m];
    v8h z8 = {};
    const v16h* wv = (const v16h*)wf1;
    for (int rt = 0; rt < 2; ++rt) {
        int ry = wave * 2 + rt;
        v8f acc = {bias, bias, bias, bias, bias, bias, bias, bias};

        // per-lane LDS base for (ky=0,kx=0); per-position byte deltas are
        // wave-uniform scalars maintained incrementally (step 2 positions):
        //   +2*CPB per kx+=2, +(22-7)*CPB extra when kx wraps past 7
        const char* apbase = (const char*)&tile[(ry * 22 + m) * CPAD + 8 * hl];
        auto loadAd = [&](int d, v8h& p0, v8h& p1, v8h& p2) {
            const _Float16* ap = (const _Float16*)(apbase + d);
            p0 = *(const v8h*)ap;
            p1 = *(const v8h*)(ap + 16);
            p2 = *(const v8h*)(ap + 32);
        };
        auto loadB = [&](int pos, v16h& q0, v16h& q1) {
            q0 = wv[(pos * 2) * 32 + lane];
            q1 = wv[(pos * 2 + 1) * 32 + lane];
        };

        v8h xa0, xa1, xa2, ya0, ya1, ya2;
        v16h xb0, xb1, yb0, yb1;
        loadAd(0, xa0, xa1, xa2);              // pos 0
        loadB(0, xb0, xb1);
        int kxX = 0, dX = 0;                   // X stream: even positions
        int kxY = 1, dY = CPB;                 // Y stream: odd positions
        #pragma unroll 1
        for (int pos = 0; pos < 48; pos += 2) {
            loadAd(dY, ya0, ya1, ya2);         // pos+1
            kxY += 2; dY += 2 * CPB;
            if (kxY >= 7) { kxY -= 7; dY += 15 * CPB; }
            loadB(pos + 1, yb0, yb1);
            acc = wmma_f16(SHUF16(xa0, xa1), xb0, acc);   // c 0..31
            acc = wmma_f16(SHUF16(xa2, z8), xb1, acc);    // c 32..47 (+0-pad)
            __builtin_amdgcn_sched_barrier(0);
            kxX += 2; dX += 2 * CPB;
            if (kxX >= 7) { kxX -= 7; dX += 15 * CPB; }
            loadAd(dX, xa0, xa1, xa2);         // pos+2 (<= 48, in range)
            loadB(pos + 2, xb0, xb1);
            acc = wmma_f16(SHUF16(ya0, ya1), yb0, acc);
            acc = wmma_f16(SHUF16(ya2, z8), yb1, acc);
            __builtin_amdgcn_sched_barrier(0);
        }
        // tail: position 48 sits in the X set
        acc = wmma_f16(SHUF16(xa0, xa1), xb0, acc);
        acc = wmma_f16(SHUF16(xa2, z8), xb1, acc);

        // D: lane channel = m, pixels M = 8*hl + r -> contiguous x
        v8h hv;
        #pragma unroll
        for (int r = 0; r < 8; ++r) hv[r] = (_Float16)acc[r];
        int y = ty0 + ry, xo = tx0 + 8 * hl;
        *(v8h*)&hbuf[(((size_t)(n * MIDC + m)) << 18) + (y << 9) + xo] = hv;
    }
}

// ---------------------------------------------------------------------------
// K3: 7x7 conv 16 -> 4 (N padded to 16) via WMMA; one K-chunk = 2 positions.
// ---------------------------------------------------------------------------
__global__ __launch_bounds__(256) __attribute__((amdgpu_waves_per_eu(4)))
void k_conv2(const _Float16* __restrict__ hbuf, const _Float16* __restrict__ wf2,
             const float* __restrict__ b2, float* __restrict__ corr4) {
    __shared__ __align__(16) _Float16 tile[22 * 22 * 16];
    int bx = blockIdx.x;
    int tx0 = (bx & 31) << 4;
    int ty0 = ((bx >> 5) & 31) << 4;
    int n = bx >> 10;
    for (int flat = threadIdx.x; flat < 16 * 484; flat += 256) {
        int c = flat / 484;
        int r = flat - c * 484;
        int dy = r / 22, dx = r - dy * 22;
        int gy = ty0 + dy - 3, gx = tx0 + dx - 3;
        _Float16 v = (_Float16)0.f;
        if ((unsigned)gy < (unsigned)HH && (unsigned)gx < (unsigned)WW)
            v = hbuf[(((size_t)(n * MIDC + c)) << 18) + (gy << 9) + gx];
        tile[(dy * 22 + dx) * 16 + c] = v;
    }
    __syncthreads();
    int wave = threadIdx.x >> 5;
    int lane = threadIdx.x & 31;
    int hl = lane >> 4, m = lane & 15;
    float bias = (m < 4) ? b2[m] : 0.f;
    const v16h* wv = (const v16h*)wf2;
    for (int rt = 0; rt < 2; ++rt) {
        int ry = wave * 2 + rt;
        v8f acc = {bias, bias, bias, bias, bias, bias, bias, bias};

        auto loadA = [&](int kc, v8h& p0, v8h& p1) {
            int pos0 = kc * 2;
            int pos1 = (pos0 + 1 < 49) ? pos0 + 1 : 48;  // clamp; 0-B kills it
            int ky0 = pos0 / 7, kx0 = pos0 - ky0 * 7;
            int ky1 = pos1 / 7, kx1 = pos1 - ky1 * 7;
            p0 = *(const v8h*)&tile[((ry + ky0) * 22 + (m + kx0)) * 16 + 8 * hl];
            p1 = *(const v8h*)&tile[((ry + ky1) * 22 + (m + kx1)) * 16 + 8 * hl];
        };

        v8h xa0, xa1, ya0, ya1;
        v16h xb, yb;
        loadA(0, xa0, xa1);
        xb = wv[lane];
        #pragma unroll 1
        for (int kc = 0; kc < 24; kc += 2) {
            loadA(kc + 1, ya0, ya1);
            yb = wv[(kc + 1) * 32 + lane];
            acc = wmma_f16(SHUF16(xa0, xa1), xb, acc);
            __builtin_amdgcn_sched_barrier(0);
            loadA(kc + 2, xa0, xa1);                     // kc+2 <= 24, in range
            xb = wv[(kc + 2) * 32 + lane];
            acc = wmma_f16(SHUF16(ya0, ya1), yb, acc);
            __builtin_amdgcn_sched_barrier(0);
        }
        // tail: chunk 24 sits in the X set
        acc = wmma_f16(SHUF16(xa0, xa1), xb, acc);

        if (m < 4) {
            int y = ty0 + ry, xo = tx0 + 8 * hl;
            float* dst = corr4 + (((size_t)(n * 4 + m)) << 18) + (y << 9) + xo;
            #pragma unroll
            for (int r = 0; r < 8; ++r) dst[r] = acc[r];
        }
    }
}

// ---------------------------------------------------------------------------
// K4: flat completion channel + rule_act add -> corr_complete, final.
// ---------------------------------------------------------------------------
__global__ void k_epilog(const float* __restrict__ corr4,
                         const unsigned char* __restrict__ rc8,
                         const float* __restrict__ ae,
                         float* __restrict__ finalp, float* __restrict__ ccomp) {
    int p = blockIdx.x * 256 + threadIdx.x;
    if (p >= NIMG * HWP) return;
    int n = p >> 18;
    int yx = p & (HWP - 1);
    float c0 = corr4[(((size_t)(n * 4 + 0)) << 18) + yx];
    float c1 = corr4[(((size_t)(n * 4 + 1)) << 18) + yx];
    float c2 = corr4[(((size_t)(n * 4 + 2)) << 18) + yx];
    float c3 = corr4[(((size_t)(n * 4 + 3)) << 18) + yx];
    float cc[5] = {c0, c1, c2, c3, -(c0 + c1 + c2 + c3)};
    int rc = rc8[p];
    #pragma unroll
    for (int ch = 0; ch < 5; ++ch) {
        float v = cc[ch];
        ccomp[(((size_t)(n * 5 + ch)) << 18) + yx] = v;
        finalp[(((size_t)(n * 5 + ch)) << 18) + yx] = ae[rc * 5 + ch] + v;
    }
}

extern "C" void kernel_launch(void* const* d_in, const int* in_sizes, int n_in,
                              void* d_out, int out_size, void* d_ws, size_t ws_size,
                              hipStream_t stream) {
    (void)in_sizes; (void)n_in; (void)out_size; (void)ws_size;
    const float* xi   = (const float*)d_in[0];
    const float* xc   = (const float*)d_in[1];
    const float* wi   = (const float*)d_in[2];
    const float* bi   = (const float*)d_in[3];
    const float* w1   = (const float*)d_in[4];
    const float* b1   = (const float*)d_in[5];
    const float* w2   = (const float*)d_in[6];
    const float* b2   = (const float*)d_in[7];
    const float* thr0 = (const float*)d_in[8];
    const float* thr1 = (const float*)d_in[9];
    const int*   rules = (const int*)d_in[10];
    const float* ae   = (const float*)d_in[11];

    char* ws = (char*)d_ws;
    _Float16* wf1 = (_Float16*)(ws + OFF_WF1);
    _Float16* wf2 = (_Float16*)(ws + OFF_WF2);
    unsigned char* rc8 = (unsigned char*)(ws + OFF_RC8);
    _Float16* hbuf = (_Float16*)(ws + OFF_H);
    float* corr4 = (float*)(ws + OFF_CORR);

    float* outp   = (float*)d_out;
    float* finalp = outp;                 // [4,5,512,512]
    float* rcat   = outp + 5242880;       // [4,512,512]
    float* ccomp  = outp + 6291456;       // [4,5,512,512]
    float* procp  = outp + 11534336;      // [4,2,512,512]

    k_pack  <<<(98 * 512 + 25 * 512 + 255) / 256, 256, 0, stream>>>(w1, w2, wf1, wf2);
    k_prolog<<<(NIMG * HWP) / 256, 256, 0, stream>>>(xi, wi, bi, thr0, thr1, rules,
                                                     procp, rcat, rc8);
    k_conv1 <<<4096, 256, 0, stream>>>(xc, ae, rc8, wf1, b1, hbuf);
    k_conv2 <<<4096, 256, 0, stream>>>(hbuf, wf2, b2, corr4);
    k_epilog<<<(NIMG * HWP) / 256, 256, 0, stream>>>(corr4, rc8, ae, finalp, ccomp);
}
